// SelfAttention_8718783610896
// MI455X (gfx1250) — matrix operand order repaired and checked by reading, not compiled, not run on hardware
//
#include <hip/hip_runtime.h>
#include <stdint.h>

#define DEVINL __device__ __forceinline__

typedef __attribute__((ext_vector_type(16))) __bf16          v16bf;
typedef __attribute__((ext_vector_type(16))) unsigned short  v16u;
typedef __attribute__((ext_vector_type(8)))  unsigned short  v8u;
typedef __attribute__((ext_vector_type(8)))  float           v8f;
typedef __attribute__((ext_vector_type(4)))  float           v4f;
typedef __attribute__((ext_vector_type(4)))  unsigned int    v4i;

constexpr int Bc = 8;
constexpr int Lc = 2048;
constexpr int Hc = 256;

DEVINL unsigned short f2bf(float f) {
  union { float f; uint32_t u; } c; c.f = f;
  uint32_t u = c.u + 0x7FFFu + ((c.u >> 16) & 1u);
  return (unsigned short)(u >> 16);
}

// pack two f32 -> two bf16 in one dword: rounding add + v_perm_b32
DEVINL unsigned int pkbf(float a, float b) {
  union { float f; uint32_t u; } ca, cb;
  ca.f = a; cb.f = b;
  uint32_t ua = ca.u + 0x7FFFu + ((ca.u >> 16) & 1u);
  uint32_t ub = cb.u + 0x7FFFu + ((cb.u >> 16) & 1u);
  return __builtin_amdgcn_perm(ub, ua, 0x07060302u);
}

DEVINL v16bf pack2(v8u lo, v8u hi) {   // elements 0..7 = lo, 8..15 = hi
  v16u u;
  #pragma unroll
  for (int i = 0; i < 8; ++i) { u[i] = lo[i]; u[i + 8] = hi[i]; }
  return __builtin_bit_cast(v16bf, u);
}

DEVINL float rmax16(float v) {
  #pragma unroll
  for (int m = 1; m < 16; m <<= 1) v = fmaxf(v, __shfl_xor(v, m, 32));
  return v;
}
DEVINL float rsum16(float v) {
  #pragma unroll
  for (int m = 1; m < 16; m <<= 1) v += __shfl_xor(v, m, 32);
  return v;
}

DEVINL v8f wmma_bf16(v16bf a, v16bf b, v8f c) {
  return __builtin_amdgcn_wmma_f32_16x16x32_bf16(false, a, false, b,
                                                 (short)0, c, false, false);
}

// ---- gfx1250 async global->LDS copy (ASYNCcnt path), with safe fallback ----
typedef int v4i32 __attribute__((vector_size(16)));

#if __has_builtin(__builtin_amdgcn_global_load_async_to_lds_b128)
#define HAVE_ASYNC_LDS 1
DEVINL void cp16_async(const unsigned short* g, unsigned short* l) {
  __builtin_amdgcn_global_load_async_to_lds_b128(
      (__attribute__((address_space(1))) v4i32*)(void*)const_cast<unsigned short*>(g),
      (__attribute__((address_space(3))) v4i32*)(void*)l,
      0, 0);
}
#else
#define HAVE_ASYNC_LDS 0
DEVINL void cp16_async(const unsigned short* g, unsigned short* l) {
  *(v8u*)l = *(const v8u*)g;     // sync fallback: b128 load + ds_store_b128
}
#endif

// ---------------------------------------------------------------------------
// Stage 0: one-pass f32 -> bf16 conversion (8 elements / thread, b128 I/O).
// ---------------------------------------------------------------------------
__global__ __launch_bounds__(256)
void cvt_kernel(const float* __restrict__ src, unsigned short* __restrict__ dst,
                int n8) {
  int i = blockIdx.x * blockDim.x + threadIdx.x;
  if (i >= n8) return;
  const v4f* p = (const v4f*)src + (size_t)i * 2;
  v4f a = p[0];
  v4f b = p[1];
  v4i o;
  o.x = pkbf(a[0], a[1]);
  o.y = pkbf(a[2], a[3]);
  o.z = pkbf(b[0], b[1]);
  o.w = pkbf(b[2], b[3]);
  *((v4i*)dst + i) = o;
}

// ---------------------------------------------------------------------------
// Stage 1: Q = x Wq^T, K = x Wk^T (bf16 row-major [B*L, H]),
//          V = x Wv^T stored transposed as Vt bf16 [B, H, L].
// ---------------------------------------------------------------------------
__global__ __launch_bounds__(256)
void proj_kernel(const unsigned short* __restrict__ xb,
                 const unsigned short* __restrict__ Wqb,
                 const unsigned short* __restrict__ Wkb,
                 const unsigned short* __restrict__ Wvb,
                 unsigned short* __restrict__ Qb,
                 unsigned short* __restrict__ Kb,
                 unsigned short* __restrict__ Vt) {
  const int ln  = threadIdx.x & 31;
  const int wid = blockIdx.x * (blockDim.x >> 5) + (threadIdx.x >> 5);
  const int NT  = Hc / 16;               // 16 n-tiles
  const int MT  = (Bc * Lc) / 16;        // 1024 m-tiles
  const int which = wid / (MT * NT);     // 0=Q 1=K 2=V
  const int rem   = wid % (MT * NT);
  const int mtile = rem / NT;
  const int ntile = rem % NT;
  const unsigned short* Wb = (which == 0) ? Wqb : ((which == 1) ? Wkb : Wvb);

  const int m    = ln & 15;
  const int offA = (ln < 16) ? 0 : 8;
  const int offB = (ln < 16) ? 0 : 8;
  const unsigned short* arow = xb + (size_t)(mtile * 16 + m) * Hc + offA;
  const unsigned short* wrow = Wb + (size_t)(ntile * 16 + m) * Hc + offB;

  v8f acc = {};
  #pragma unroll
  for (int c = 0; c < 8; ++c) {          // K chunks of 32
    v8u a0 = *(const v8u*)(arow + c * 32);
    v8u a1 = *(const v8u*)(arow + c * 32 + 16);
    v8u b0 = *(const v8u*)(wrow + c * 32);
    v8u b1 = *(const v8u*)(wrow + c * 32 + 16);
    acc = wmma_bf16(pack2(a0, a1), pack2(b0, b1), acc);
  }

  const int n      = ln & 15;
  const int mh     = (ln < 16) ? 0 : 8;
  const int token0 = mtile * 16;
  if (which == 2) {
    const int b  = token0 / Lc;
    const int l0 = token0 - b * Lc + mh;
    const int o  = ntile * 16 + n;
    v4i pk;
    pk.x = pkbf(acc[0], acc[1]);
    pk.y = pkbf(acc[2], acc[3]);
    pk.z = pkbf(acc[4], acc[5]);
    pk.w = pkbf(acc[6], acc[7]);
    *(v4i*)(Vt + ((size_t)(b * Hc + o)) * Lc + l0) = pk;
  } else {
    unsigned short* dst = (which == 0) ? Qb : Kb;
    #pragma unroll
    for (int r = 0; r < 8; ++r)
      dst[(size_t)(token0 + mh + r) * Hc + ntile * 16 + n] = f2bf(acc[r]);
  }
}

// ---------------------------------------------------------------------------
// Stage 2: flash attention, workgroup-cooperative K/V staging.
// 8 waves / block share batch b: K-block (32x256) and V-block (256x32) are
// async-copied to LDS once per block per iteration; each wave computes its
// own 16-query tile from LDS.
// ---------------------------------------------------------------------------
__global__ __launch_bounds__(256)
void flash_kernel(const unsigned short* __restrict__ Qb,
                  const unsigned short* __restrict__ Kb,
                  const unsigned short* __restrict__ Vt,
                  const int* __restrict__ lens,
                  float* __restrict__ out) {
  __shared__ __align__(16) unsigned short Ksh[32 * 256];   // 16 KB
  __shared__ __align__(16) unsigned short Vsh[256 * 32];   // 16 KB
  __shared__ __align__(16) unsigned short Psh[8][16 * 32]; //  8 KB

  const int tid   = threadIdx.x;
  const int ln    = tid & 31;
  const int warp  = tid >> 5;
  const int wid   = blockIdx.x * 8 + warp;
  const int b     = wid / (Lc / 16);      // uniform across the block
  const int qtile = wid % (Lc / 16);
  const int q0    = qtile * 16;
  unsigned short* slab = &Psh[warp][0];

  const int m    = ln & 15;
  const int offA = (ln < 16) ? 0 : 8;
  const int offB = (ln < 16) ? 0 : 8;

  // --- load Q tile (16 x 256) into registers, A-operand layout ---
  v16bf qa[8];
  {
    const unsigned short* qrow = Qb + ((size_t)(b * Lc + q0 + m)) * Hc;
    #pragma unroll
    for (int c = 0; c < 8; ++c) {
      v8u h0 = *(const v8u*)(qrow + c * 32 + offA);
      v8u h1 = *(const v8u*)(qrow + c * 32 + 16 + offA);
      qa[c] = pack2(h0, h1);
    }
  }

  v8f O[16] = {};
  float mrow[8], lrow[8];
  #pragma unroll
  for (int r = 0; r < 8; ++r) { mrow[r] = -1e30f; lrow[r] = 0.f; }

  const int   len   = lens[b];            // uniform across the block
  const int   nb    = (len + 31) >> 5;
  const float scale = 0.0625f;            // 1/sqrt(256)

  for (int j = 0; j < nb; ++j) {
    const int kbase = j * 32;

    // ---- cooperative async copy of K-block and V-block into LDS ----
    // K: 32 rows x 512B = 1024 x 16B chunks; V: 256 rows x 64B = 1024 chunks
    #pragma unroll
    for (int i = 0; i < 4; ++i) {
      const int ch  = tid + i * 256;
      const int row = ch >> 5;
      const int c8  = (ch & 31) * 8;
      cp16_async(Kb + ((size_t)(b * Lc + kbase + row)) * Hc + c8, Ksh + ch * 8);
    }
    #pragma unroll
    for (int i = 0; i < 4; ++i) {
      const int ch  = tid + i * 256;
      const int row = ch >> 2;
      const int c8  = (ch & 3) * 8;
      cp16_async(Vt + ((size_t)(b * Hc + row)) * Lc + kbase + c8, Vsh + ch * 8);
    }
#if HAVE_ASYNC_LDS
    asm volatile("s_wait_asynccnt 0" ::: "memory");
#endif
    __syncthreads();

    if (j + 1 < nb) {   // prefetch next K block (global_prefetch_b8)
      __builtin_prefetch(Kb + ((size_t)(b * Lc + kbase + 32 + (tid >> 3))) * Hc, 0, 1);
    }

    // ---- S = Q K^T (B operand from LDS) ----
    v8f s[2];
    #pragma unroll
    for (int t = 0; t < 2; ++t) {
      v8f acc = {};
      const unsigned short* kr = Ksh + (t * 16 + m) * Hc + offB;
      #pragma unroll
      for (int c = 0; c < 8; ++c) {
        v8u h0 = *(const v8u*)(kr + c * 32);
        v8u h1 = *(const v8u*)(kr + c * 32 + 16);
        acc = wmma_bf16(qa[c], pack2(h0, h1), acc);
      }
      s[t] = acc;
    }

    // ---- online softmax over the 32 columns of this block ----
    const int  col0 = kbase + m;
    const bool v0   = col0 < len;
    const bool v1   = (col0 + 16) < len;
    #pragma unroll
    for (int r = 0; r < 8; ++r) {
      float s0 = v0 ? s[0][r] * scale : -1e30f;
      float s1 = v1 ? s[1][r] * scale : -1e30f;
      float rm   = rmax16(fmaxf(s0, s1));
      float newm = fmaxf(mrow[r], rm);
      float e0 = v0 ? __expf(s0 - newm) : 0.f;
      float e1 = v1 ? __expf(s1 - newm) : 0.f;
      float rs = rsum16(e0 + e1);
      float cf = __expf(mrow[r] - newm);
      lrow[r] = lrow[r] * cf + rs;
      mrow[r] = newm;
      #pragma unroll
      for (int t = 0; t < 16; ++t) O[t][r] *= cf;
      const int mm = r + ((ln < 16) ? 0 : 8);
      slab[mm * 32 + m]      = f2bf(e0);
      slab[mm * 32 + 16 + m] = f2bf(e1);
    }

    asm volatile("s_wait_dscnt 0" ::: "memory");

    // ---- re-load P in A-operand layout (16 x 32) ----
    v16bf pa;
    {
      const unsigned short* pr = slab + m * 32;
      v8u h0 = *(const v8u*)(pr + offA);
      v8u h1 = *(const v8u*)(pr + 16 + offA);
      pa = pack2(h0, h1);
    }

    // ---- O += P V (B operand from LDS V^T block) ----
    #pragma unroll
    for (int t = 0; t < 16; ++t) {
      const unsigned short* vr = Vsh + (t * 16 + m) * 32 + offB;
      v8u h0 = *(const v8u*)vr;
      v8u h1 = *(const v8u*)(vr + 16);
      O[t] = wmma_bf16(pa, pack2(h0, h1), O[t]);
    }

    __syncthreads();   // protect LDS K/V before next iteration overwrites
  }

  // ---- normalize and store fp32 output ----
  const int mh = (ln < 16) ? 0 : 8;
  float inv[8];
  #pragma unroll
  for (int r = 0; r < 8; ++r) inv[r] = 1.0f / lrow[r];
  #pragma unroll
  for (int t = 0; t < 16; ++t) {
    #pragma unroll
    for (int r = 0; r < 8; ++r) {
      out[((size_t)(b * Lc + q0 + mh + r)) * Hc + t * 16 + m] = O[t][r] * inv[r];
    }
  }
}

// ---------------------------------------------------------------------------
extern "C" void kernel_launch(void* const* d_in, const int* in_sizes, int n_in,
                              void* d_out, int out_size, void* d_ws, size_t ws_size,
                              hipStream_t stream) {
  (void)in_sizes; (void)n_in; (void)out_size; (void)ws_size;
  const float* x   = (const float*)d_in[0];
  const float* Wq  = (const float*)d_in[1];
  const float* Wk  = (const float*)d_in[2];
  const float* Wv  = (const float*)d_in[3];
  const int*   len = (const int*)d_in[4];
  float*       out = (float*)d_out;

  const size_t xelems = (size_t)Bc * Lc * Hc;    // 4M
  const size_t welems = (size_t)Hc * Hc;         // 64K
  unsigned short* xb  = (unsigned short*)d_ws;   // 8 MB
  unsigned short* Wqb = xb  + xelems;
  unsigned short* Wkb = Wqb + welems;
  unsigned short* Wvb = Wkb + welems;
  unsigned short* Qb  = Wvb + welems;            // 8 MB
  unsigned short* Kb  = Qb  + xelems;            // 8 MB
  unsigned short* Vt  = Kb  + xelems;            // 8 MB

  cvt_kernel<<<dim3((int)(xelems / 8 / 256)), dim3(256), 0, stream>>>(x, xb, (int)(xelems / 8));
  cvt_kernel<<<dim3((int)(welems / 8 / 256)), dim3(256), 0, stream>>>(Wq, Wqb, (int)(welems / 8));
  cvt_kernel<<<dim3((int)(welems / 8 / 256)), dim3(256), 0, stream>>>(Wk, Wkb, (int)(welems / 8));
  cvt_kernel<<<dim3((int)(welems / 8 / 256)), dim3(256), 0, stream>>>(Wv, Wvb, (int)(welems / 8));

  proj_kernel<<<dim3(6144), dim3(256), 0, stream>>>(xb, Wqb, Wkb, Wvb, Qb, Kb, Vt);
  flash_kernel<<<dim3(128), dim3(256), 0, stream>>>(Qb, Kb, Vt, len, out);
}
